// AutoregressivePolicy_60327110640324
// MI455X (gfx1250) — matrix operand order repaired and checked by reading, not compiled, not run on hardware
//
#include <hip/hip_runtime.h>
#include <math.h>

typedef __bf16 bf16;
typedef __bf16 v16bf __attribute__((ext_vector_type(16)));
typedef __bf16 v8bf  __attribute__((ext_vector_type(8)));
typedef float  v8f   __attribute__((ext_vector_type(8)));

#define PI_F    3.14159265358979323846f
#define TWOPI_F 6.28318530717958647692f

// ---------------------------------------------------------------------------
// WMMA fragment loaders (CDNA5 16x16x32 bf16 layouts, CDNA5 ISA)
// ---------------------------------------------------------------------------

// A[MxK] row-major (LDS): per lane = row m, two contiguous 8-elem runs.
static __device__ __forceinline__ v16bf wmma_load_a(const bf16* __restrict__ A, int lda,
                                                    int m0, int k0, int lane) {
  const int m    = m0 + (lane & 15);
  const int half = ((lane >> 4) << 3);          // lanes 16..31 hold K+8 block
  const bf16* p  = A + m * lda + k0 + half;
  v8bf lo = *(const v8bf*)p;                    // k0+half   .. +8   -> r[0..7]
  v8bf hi = *(const v8bf*)(p + 16);             // k0+16+half.. +8   -> r[8..15]
  return __builtin_shufflevector(lo, hi, 0, 1, 2, 3, 4, 5, 6, 7,
                                 8, 9, 10, 11, 12, 13, 14, 15);
}

// A fragment for a 1-row "vector" operand: row 0 = vec, rows 1..15 = 0.
static __device__ __forceinline__ v16bf wmma_load_a_vec(const bf16* __restrict__ vec,
                                                        int k0, int lane) {
  v16bf r;
#pragma unroll
  for (int j = 0; j < 16; ++j) r[j] = (bf16)0.f;
  if ((lane & 15) == 0) {
    const int half = ((lane >> 4) << 3);
    v8bf lo = *(const v8bf*)(vec + k0 + half);
    v8bf hi = *(const v8bf*)(vec + k0 + 16 + half);
    r = __builtin_shufflevector(lo, hi, 0, 1, 2, 3, 4, 5, 6, 7,
                                8, 9, 10, 11, 12, 13, 14, 15);
  }
  return r;
}

// B stored pre-swizzled in fragment order: 32B contiguous per lane per tile.
static __device__ __forceinline__ v16bf wmma_load_bs(const bf16* __restrict__ Bs,
                                                     int kt, int k0, int n0, int lane) {
  const bf16* p = Bs + (size_t)((((n0 >> 4) * kt + (k0 >> 5)) << 5) + lane) * 16;
  return *(const v16bf*)p;
}

// C[MxN](bf16) = epilogue(A[MxK] @ Bs + bias); waves split 16x16 tiles.
static __device__ void mm_bf16(const bf16* __restrict__ A, int lda,
                               const bf16* __restrict__ Bs,
                               const float* __restrict__ bias,
                               bf16* __restrict__ C, int ldc,
                               int M, int N, int K, bool relu,
                               const float* __restrict__ rowmul) {
  const int lane  = threadIdx.x & 31;
  const int wave  = threadIdx.x >> 5;
  const int nw    = blockDim.x >> 5;
  const int kt    = K >> 5;
  const int nt    = N >> 4;
  const int ntile = (M >> 4) * nt;
  for (int tile = wave; tile < ntile; tile += nw) {
    const int m0 = (tile / nt) << 4;
    const int n0 = (tile % nt) << 4;
    v8f acc = {};
    for (int k0 = 0; k0 < K; k0 += 32) {
      v16bf a = wmma_load_a(A, lda, m0, k0, lane);
      v16bf b = wmma_load_bs(Bs, kt, k0, n0, lane);
      acc = __builtin_amdgcn_wmma_f32_16x16x32_bf16(false, a, false, b,
                                                    (short)0, acc, false, false);
    }
    const int n  = n0 + (lane & 15);
    const int mb = m0 + ((lane >> 4) << 3);
    const float bn = bias ? bias[n] : 0.f;
#pragma unroll
    for (int j = 0; j < 8; ++j) {
      float v = acc[j] + bn;
      if (relu)   v = fmaxf(v, 0.f);
      if (rowmul) v *= rowmul[mb + j];
      C[(mb + j) * ldc + n] = (bf16)v;
    }
  }
}

// out[1xN] = vec[1xK] @ Bs[KxN] + bias, via WMMA with M=1 padded to 16.
// mode: 0 = none, 1 = relu, 2 = exact gelu. out (f32) and outb (bf16) optional.
static __device__ void mv_wmma(const bf16* __restrict__ vecb,
                               const bf16* __restrict__ Bs,
                               const float* __restrict__ bias,
                               float* __restrict__ out, bf16* __restrict__ outb,
                               int N, int K, int mode) {
  const int lane = threadIdx.x & 31;
  const int wave = threadIdx.x >> 5;
  const int nw   = blockDim.x >> 5;
  const int kt   = K >> 5;
  for (int in = wave; in < (N >> 4); in += nw) {
    v8f acc = {};
    for (int k0 = 0; k0 < K; k0 += 32) {
      v16bf a = wmma_load_a_vec(vecb, k0, lane);
      v16bf b = wmma_load_bs(Bs, kt, k0, in << 4, lane);
      acc = __builtin_amdgcn_wmma_f32_16x16x32_bf16(false, a, false, b,
                                                    (short)0, acc, false, false);
    }
    if (lane < 16) {                       // D row 0: VGPR0, lanes 0..15, n = lane
      const int n = (in << 4) + lane;
      float v = acc[0] + (bias ? bias[n] : 0.f);
      if (mode == 1)      v = fmaxf(v, 0.f);
      else if (mode == 2) v = 0.5f * v * (1.f + erff(v * 0.70710678118654752f));
      if (out)  out[n]  = v;
      if (outb) outb[n] = (bf16)v;
    }
  }
}

// ---------------------------------------------------------------------------
// Prep: swizzle weights (w: (dout,din) row-major) into WMMA-B fragment order.
// Layout: tile (in = n0/16 outer, ik = k0/32 inner) -> 32 lanes -> 16 elems,
// with k = ik*32 + (lane>=16)*8 + (e&7) + (e>=8)*16, n = in*16 + (lane&15).
// ---------------------------------------------------------------------------
__global__ void swizzle_b_kernel(const float* __restrict__ w, bf16* __restrict__ bs,
                                 int din, int dout, int kpad) {
  int i = blockIdx.x * blockDim.x + threadIdx.x;
  int total = kpad * dout;
  if (i >= total) return;
  int e    = i & 15;
  int lane = (i >> 4) & 31;
  int tile = i >> 9;
  int kt   = kpad >> 5;
  int ik   = tile % kt;
  int in   = tile / kt;
  int k = (ik << 5) + ((lane >> 4) << 3) + (e & 7) + ((e >> 3) << 4);
  int n = (in << 4) + (lane & 15);
  bs[i] = (k < din) ? (bf16)w[n * din + k] : (bf16)0.f;
}

// ---------------------------------------------------------------------------
// Map lane encoder: per (b,lane) pointnet 3->32->64->128, maxpool(20), mask,
// then lane_proj 128->256 -> bf16 map_proj (one-time cost, VALU is fine)
// ---------------------------------------------------------------------------
__global__ __launch_bounds__(128)
void map_encode_kernel(const float* __restrict__ lanes, const float* __restrict__ lmask,
                       const float* __restrict__ le1w, const float* __restrict__ le1b,
                       const float* __restrict__ le2w, const float* __restrict__ le2b,
                       const float* __restrict__ le3w, const float* __restrict__ le3b,
                       const float* __restrict__ lpw,  const float* __restrict__ lpb,
                       bf16* __restrict__ mapproj) {
  const int bl = blockIdx.x;
  const int b = bl >> 6, ln = bl & 63;
  __shared__ float pts[20 * 3];
  __shared__ float o1[20 * 32];
  __shared__ float o2[20 * 64];
  __shared__ float o3[20 * 128];
  __shared__ float lf[128];

  const float* src = lanes + ((size_t)(b * 64 + ln)) * 60;
  for (int i = threadIdx.x; i < 60; i += 128) pts[i] = src[i];
  __syncthreads();

  for (int i = threadIdx.x; i < 640; i += 128) {           // 20x32, relu
    int p = i >> 5, c = i & 31;
    float acc = le1b[c];
#pragma unroll
    for (int k = 0; k < 3; ++k) acc += pts[p * 3 + k] * le1w[c * 3 + k];
    o1[i] = fmaxf(acc, 0.f);
  }
  __syncthreads();
  for (int i = threadIdx.x; i < 1280; i += 128) {          // 20x64, relu
    int p = i >> 6, c = i & 63;
    float acc = le2b[c];
    for (int k = 0; k < 32; ++k) acc += o1[p * 32 + k] * le2w[c * 32 + k];
    o2[i] = fmaxf(acc, 0.f);
  }
  __syncthreads();
  for (int i = threadIdx.x; i < 2560; i += 128) {          // 20x128
    int p = i >> 7, c = i & 127;
    float acc = le3b[c];
    for (int k = 0; k < 64; ++k) acc += o2[p * 64 + k] * le3w[c * 64 + k];
    o3[i] = acc;
  }
  __syncthreads();
  const float m = lmask[b * 64 + ln];
  for (int c = threadIdx.x; c < 128; c += 128) {           // maxpool over 20 pts
    float mx = o3[c];
    for (int p = 1; p < 20; ++p) mx = fmaxf(mx, o3[p * 128 + c]);
    lf[c] = mx * m;
  }
  __syncthreads();
  bf16* dst = mapproj + ((size_t)(b * 64 + ln)) * 256;
  for (int n = threadIdx.x; n < 256; n += 128) {           // lane_proj 128->256
    float acc = lpb[n];
    for (int k = 0; k < 128; ++k) acc += lf[k] * lpw[n * 128 + k];
    dst[n] = (bf16)acc;
  }
}

// Step-invariant map K/V: per b, [64x256] @ wk^T/wv^T via WMMA
__global__ __launch_bounds__(256)
void map_kv_kernel(const bf16* __restrict__ mapproj,
                   const bf16* __restrict__ WKs, const bf16* __restrict__ WVs,
                   const float* __restrict__ bk, const float* __restrict__ bv,
                   bf16* __restrict__ mapk, bf16* __restrict__ mapv) {
  const int b = blockIdx.x;
  const bf16* A = mapproj + (size_t)b * 64 * 256;
  mm_bf16(A, 256, WKs, bk, mapk + (size_t)b * 64 * 256, 256, 64, 256, 256, false, nullptr);
  mm_bf16(A, 256, WVs, bv, mapv + (size_t)b * 64 * 256, 256, 64, 256, 256, false, nullptr);
}

// ---------------------------------------------------------------------------
// Main autoregressive scan: one block (8 wave32) per batch element, 80 steps
// ---------------------------------------------------------------------------
struct StepParams {
  const float* agents_seq;
  const float* agents_mask;
  const int*   mode_c;
  const float* mode_embed;
  const bf16 *W1s, *W2s, *W3s, *WKs, *WVs, *WQs, *WOs, *F1s, *F2s, *A1s;
  const float *b1, *b2, *b3, *bk, *bv, *bq, *bo;
  const float *g1, *be1, *g2, *be2;
  const float *fb1, *fb2, *ab1, *act2w, *ab2;
  const bf16  *mapk, *mapv;
  float* out;
};

static __device__ __forceinline__ float block_sum256(float v, float* red, int tid) {
#pragma unroll
  for (int off = 16; off; off >>= 1) v += __shfl_xor(v, off, 32);
  if ((tid & 31) == 0) red[tid >> 5] = v;
  __syncthreads();
  float s = 0.f;
#pragma unroll
  for (int i = 0; i < 8; ++i) s += red[i];
  __syncthreads();
  return s;
}

__global__ __launch_bounds__(256)
void policy_scan_kernel(StepParams P) {
  const int b    = blockIdx.x;
  const int tid  = threadIdx.x;
  const int lane = tid & 31;
  const int wave = tid >> 5;

  __shared__ __align__(32) bf16  bufA[16384]; // featb[32x32]/h2[32x128]/Ka,Va[2x32x256]
  __shared__ __align__(32) bf16  bufB[8192];  // h1[32x64]  /pa_sel[32x256]
  __shared__ float feat[64 * 10];
  __shared__ float dist[64];
  __shared__ int   sel[32];
  __shared__ float rowm[32];
  __shared__ float mq[256];
  __shared__ __align__(32) bf16 mqb[256];     // bf16 mirror of mq
  __shared__ float vecA[256];                 // qh / attn_out / act hidden
  __shared__ float vecB[256];                 // attn o / ffn out
  __shared__ __align__(32) bf16 vecBb[256];   // bf16 mirror of attn o
  __shared__ float vecC[256];                 // post-LN1 residual stream
  __shared__ __align__(32) bf16 vecCb[256];   // bf16 mirror
  __shared__ __align__(32) bf16 ub[1024];     // ffn hidden (bf16)
  __shared__ float sc[4 * 96];                // attention scores / probs
  __shared__ float red[8];
  __shared__ float ego[4];

  if (tid < 4) ego[tid] = 0.f;
  {
    const int mode = P.mode_c[b];
    const float m0v = P.mode_embed[mode * 256 + tid];
    mq[tid]  = m0v;
    mqb[tid] = (bf16)m0v;
  }
  __syncthreads();

  for (int t = 0; t < 80; ++t) {
    const float* ag = P.agents_seq + (((size_t)b * 80 + t) * 64) * 10;
    if (tid == 0 && t + 1 < 80)
      __builtin_prefetch(P.agents_seq + (((size_t)b * 80 + t + 1) * 64) * 10, 0, 3);

    // --- phase 0: ego-frame features (H=1 -> t_n = -1) + distances ---
    if (tid < 64) {
      const float ex = ego[0], ey = ego[1], eh = ego[2];
      const float c = cosf(eh), s = -sinf(eh);            // cos(-eh), sin(-eh)
      const float* a = ag + tid * 10;
      const float dx = a[0] - ex, dy = a[1] - ey;
      const float xe = c * dx - s * dy;
      const float ye = s * dx + c * dy;
      float r = a[2] - eh + PI_F;
      r = r - TWOPI_F * floorf(r / TWOPI_F);
      const float he = r - PI_F;
      const float vx = c * a[3] - s * a[4];
      const float vy = s * a[3] + c * a[4];
      float* f = feat + tid * 10;
      f[0] = xe; f[1] = ye; f[2] = he; f[3] = vx; f[4] = vy;
      f[5] = a[5]; f[6] = a[6]; f[7] = a[7]; f[8] = -1.f; f[9] = a[9];
      dist[tid] = sqrtf(xe * xe + ye * ye);
    }
    __syncthreads();

    // --- phase 1: exact rank-based top-KNN (matches top_k tie-breaking) ---
    if (tid < 64) {
      const float di = dist[tid];
      int rank = 0;
      for (int j = 0; j < 64; ++j) {
        const float dj = dist[j];
        rank += (dj < di) || (dj == di && j < tid);
      }
      if (rank < 32) sel[rank] = tid;
    }
    __syncthreads();
    if (tid < 32) rowm[tid] = P.agents_mask[b * 64 + sel[tid]];

    // --- phase 2: gather selected agents into bf16 A-tile (K padded to 32) ---
    bf16* featb = bufA;
    for (int i = tid; i < 32 * 32; i += 256) {
      const int r = i >> 5, k = i & 31;
      featb[i] = (k < 10) ? (bf16)feat[sel[r] * 10 + k] : (bf16)0.f;
    }
    __syncthreads();

    // --- phases 3-6: pointnet + K/V projection, all WMMA bf16 ---
    mm_bf16(featb, 32, P.W1s, P.b1, bufB, 64,  32, 64,  32,  true,  nullptr);
    __syncthreads();
    mm_bf16(bufB, 64,  P.W2s, P.b2, bufA, 128, 32, 128, 64,  true,  nullptr);
    __syncthreads();
    mm_bf16(bufA, 128, P.W3s, P.b3, bufB, 256, 32, 256, 128, false, rowm);
    __syncthreads();
    mm_bf16(bufB, 256, P.WKs, P.bk, bufA,        256, 32, 256, 256, false, nullptr);
    mm_bf16(bufB, 256, P.WVs, P.bv, bufA + 8192, 256, 32, 256, 256, false, nullptr);
    __syncthreads();

    // --- phase 7: q projection (WMMA, M=1 padded) ---
    mv_wmma(mqb, P.WQs, P.bq, vecA, nullptr, 256, 256, 0);
    __syncthreads();

    // --- phase 8: attention scores, 4 heads x 96 keys ---
    const bf16* Ka = bufA;
    const bf16* Va = bufA + 8192;
    const bf16* mk = P.mapk + (size_t)b * 64 * 256;
    const bf16* mv = P.mapv + (size_t)b * 64 * 256;
    for (int j = tid; j < 384; j += 256) {
      const int h = j / 96, key = j % 96;
      const float* q = vecA + h * 64;
      const bf16* kr = (key < 32) ? (Ka + key * 256 + h * 64)
                                  : (mk + (size_t)(key - 32) * 256 + h * 64);
      float acc = 0.f;
#pragma unroll
      for (int c = 0; c < 8; ++c) {
        v8bf kv8 = *(const v8bf*)(kr + c * 8);
#pragma unroll
        for (int d = 0; d < 8; ++d) acc += q[c * 8 + d] * (float)kv8[d];
      }
      sc[j] = acc * 0.125f;
    }
    __syncthreads();

    // --- phase 9: softmax per head (wave32 shuffle reduction) ---
    if (wave < 4) {
      const int base = wave * 96;
      float v0 = sc[base + lane], v1 = sc[base + lane + 32], v2 = sc[base + lane + 64];
      float m = fmaxf(v0, fmaxf(v1, v2));
#pragma unroll
      for (int off = 16; off; off >>= 1) m = fmaxf(m, __shfl_xor(m, off, 32));
      const float e0 = expf(v0 - m), e1 = expf(v1 - m), e2 = expf(v2 - m);
      float ssum = e0 + e1 + e2;
#pragma unroll
      for (int off = 16; off; off >>= 1) ssum += __shfl_xor(ssum, off, 32);
      const float inv = 1.f / ssum;
      sc[base + lane] = e0 * inv; sc[base + lane + 32] = e1 * inv; sc[base + lane + 64] = e2 * inv;
    }
    __syncthreads();

    // --- phase 10: weighted value sum ---
    {
      const int h = tid >> 6;
      const float* a = sc + h * 96;
      float acc = 0.f;
      for (int j = 0; j < 32; ++j) acc += a[j] * (float)Va[j * 256 + tid];
      for (int j = 0; j < 64; ++j) acc += a[32 + j] * (float)mv[(size_t)j * 256 + tid];
      vecB[tid]  = acc;
      vecBb[tid] = (bf16)acc;
    }
    __syncthreads();

    // --- phase 11: output proj (WMMA) + residual + LayerNorm1 ---
    mv_wmma(vecBb, P.WOs, P.bo, vecA, nullptr, 256, 256, 0);
    __syncthreads();
    {
      const float x = mq[tid] + vecA[tid];
      const float s1 = block_sum256(x, red, tid);
      const float s2 = block_sum256(x * x, red, tid);
      const float mu = s1 * (1.f / 256.f);
      const float var = s2 * (1.f / 256.f) - mu * mu;
      const float rstd = rsqrtf(var + 1e-5f);
      const float y = (x - mu) * rstd * P.g1[tid] + P.be1[tid];
      vecC[tid]  = y;
      vecCb[tid] = (bf16)y;
    }
    __syncthreads();

    // --- phase 12: FFN hidden = gelu(vecC @ ffn1^T + b) (WMMA) ---
    mv_wmma(vecCb, P.F1s, P.fb1, nullptr, ub, 1024, 256, 2);
    __syncthreads();

    // --- phase 13: FFN out (WMMA) + residual + LayerNorm2 -> mq ---
    mv_wmma(ub, P.F2s, P.fb2, vecB, nullptr, 256, 1024, 0);
    __syncthreads();
    {
      const float x = vecC[tid] + vecB[tid];
      const float s1 = block_sum256(x, red, tid);
      const float s2 = block_sum256(x * x, red, tid);
      const float mu = s1 * (1.f / 256.f);
      const float var = s2 * (1.f / 256.f) - mu * mu;
      const float rstd = rsqrtf(var + 1e-5f);
      const float y = (x - mu) * rstd * P.g2[tid] + P.be2[tid];
      mq[tid]  = y;
      mqb[tid] = (bf16)y;
    }
    __syncthreads();

    // --- phase 14: action head (WMMA relu) -> tiny act2 -> ego ---
    mv_wmma(mqb, P.A1s, P.ab1, vecA, nullptr, 256, 256, 1);
    __syncthreads();
    if (tid < 3) {
      float acc = P.ab2[tid];
      const float* w = P.act2w + tid * 256;
      for (int k = 0; k < 256; ++k) acc += vecA[k] * w[k];
      P.out[((size_t)b * 80 + t) * 3 + tid] = acc;
      ego[tid] = acc;
    }
    __syncthreads();
  }
}

// ---------------------------------------------------------------------------
// Host launch
// ---------------------------------------------------------------------------
static inline size_t align256(size_t x) { return (x + 255) & ~(size_t)255; }

extern "C" void kernel_launch(void* const* d_in, const int* in_sizes, int n_in,
                              void* d_out, int out_size, void* d_ws, size_t ws_size,
                              hipStream_t stream) {
  (void)in_sizes; (void)n_in; (void)out_size; (void)ws_size;

  const float* agents_seq     = (const float*)d_in[1];
  const float* agents_mask    = (const float*)d_in[2];
  const int*   mode_c         = (const int*)d_in[3];
  const float* map_lanes      = (const float*)d_in[4];
  const float* map_lanes_mask = (const float*)d_in[5];

  const float* mode_embed = (const float*)d_in[6];
  const float* pn1w = (const float*)d_in[7];   const float* pn1b = (const float*)d_in[8];
  const float* pn2w = (const float*)d_in[9];   const float* pn2b = (const float*)d_in[10];
  const float* pn3w = (const float*)d_in[11];  const float* pn3b = (const float*)d_in[12];
  const float* le1w = (const float*)d_in[13];  const float* le1b = (const float*)d_in[14];
  const float* le2w = (const float*)d_in[15];  const float* le2b = (const float*)d_in[16];
  const float* le3w = (const float*)d_in[17];  const float* le3b = (const float*)d_in[18];
  const float* wqw  = (const float*)d_in[19];  const float* wqb  = (const float*)d_in[20];
  const float* wkw  = (const float*)d_in[21];  const float* wkb  = (const float*)d_in[22];
  const float* wvw  = (const float*)d_in[23];  const float* wvb  = (const float*)d_in[24];
  const float* wow  = (const float*)d_in[25];  const float* wob  = (const float*)d_in[26];
  const float* n1g  = (const float*)d_in[27];  const float* n1b  = (const float*)d_in[28];
  const float* n2g  = (const float*)d_in[29];  const float* n2b  = (const float*)d_in[30];
  const float* f1w  = (const float*)d_in[31];  const float* f1b  = (const float*)d_in[32];
  const float* f2w  = (const float*)d_in[33];  const float* f2b  = (const float*)d_in[34];
  const float* lpw  = (const float*)d_in[35];  const float* lpb  = (const float*)d_in[36];
  const float* a1w  = (const float*)d_in[37];  const float* a1b  = (const float*)d_in[38];
  const float* a2w  = (const float*)d_in[39];  const float* a2b  = (const float*)d_in[40];

  // workspace carve
  char* ws = (char*)d_ws;
  size_t off = 0;
  auto take = [&](size_t bytes) { char* p = ws + off; off = align256(off + bytes); return p; };

  bf16* W1s = (bf16*)take(32   * 64   * 2);
  bf16* W2s = (bf16*)take(64   * 128  * 2);
  bf16* W3s = (bf16*)take(128  * 256  * 2);
  bf16* WKs = (bf16*)take(256  * 256  * 2);
  bf16* WVs = (bf16*)take(256  * 256  * 2);
  bf16* WQs = (bf16*)take(256  * 256  * 2);
  bf16* WOs = (bf16*)take(256  * 256  * 2);
  bf16* F1s = (bf16*)take(256  * 1024 * 2);
  bf16* F2s = (bf16*)take(1024 * 256  * 2);
  bf16* A1s = (bf16*)take(256  * 256  * 2);
  bf16* mapproj = (bf16*)take((size_t)64 * 64 * 256 * 2);
  bf16* mapk    = (bf16*)take((size_t)64 * 64 * 256 * 2);
  bf16* mapv    = (bf16*)take((size_t)64 * 64 * 256 * 2);

  // 1) weight swizzles into WMMA fragment order
  auto grid = [](int total) { return dim3((total + 255) / 256); };
  swizzle_b_kernel<<<grid(32 * 64),    256, 0, stream>>>(pn1w, W1s, 10,   64,   32);
  swizzle_b_kernel<<<grid(64 * 128),   256, 0, stream>>>(pn2w, W2s, 64,   128,  64);
  swizzle_b_kernel<<<grid(128 * 256),  256, 0, stream>>>(pn3w, W3s, 128,  256,  128);
  swizzle_b_kernel<<<grid(256 * 256),  256, 0, stream>>>(wkw,  WKs, 256,  256,  256);
  swizzle_b_kernel<<<grid(256 * 256),  256, 0, stream>>>(wvw,  WVs, 256,  256,  256);
  swizzle_b_kernel<<<grid(256 * 256),  256, 0, stream>>>(wqw,  WQs, 256,  256,  256);
  swizzle_b_kernel<<<grid(256 * 256),  256, 0, stream>>>(wow,  WOs, 256,  256,  256);
  swizzle_b_kernel<<<grid(256 * 1024), 256, 0, stream>>>(f1w,  F1s, 256,  1024, 256);
  swizzle_b_kernel<<<grid(1024 * 256), 256, 0, stream>>>(f2w,  F2s, 1024, 256,  1024);
  swizzle_b_kernel<<<grid(256 * 256),  256, 0, stream>>>(a1w,  A1s, 256,  256,  256);

  // 2) map encoder + step-invariant map K/V (WMMA)
  map_encode_kernel<<<dim3(64 * 64), 128, 0, stream>>>(
      map_lanes, map_lanes_mask, le1w, le1b, le2w, le2b, le3w, le3b, lpw, lpb, mapproj);
  map_kv_kernel<<<dim3(64), 256, 0, stream>>>(mapproj, WKs, WVs, wkb, wvb, mapk, mapv);

  // 3) autoregressive scan: one persistent block per batch element
  StepParams P;
  P.agents_seq = agents_seq; P.agents_mask = agents_mask; P.mode_c = mode_c;
  P.mode_embed = mode_embed;
  P.W1s = W1s; P.W2s = W2s; P.W3s = W3s; P.WKs = WKs; P.WVs = WVs;
  P.WQs = WQs; P.WOs = WOs; P.F1s = F1s; P.F2s = F2s; P.A1s = A1s;
  P.b1 = pn1b; P.b2 = pn2b; P.b3 = pn3b; P.bk = wkb; P.bv = wvb;
  P.bq = wqb; P.bo = wob;
  P.g1 = n1g; P.be1 = n1b; P.g2 = n2g; P.be2 = n2b;
  P.fb1 = f1b; P.fb2 = f2b; P.ab1 = a1b; P.act2w = a2w; P.ab2 = a2b;
  P.mapk = mapk; P.mapv = mapv;
  P.out = (float*)d_out;

  policy_scan_kernel<<<dim3(64), 256, 0, stream>>>(P);
}
